// EstNUNet_41961830482355
// MI455X (gfx1250) — compile-verified, matrix operand order checked by reading, not run on hardware
//
#include <hip/hip_runtime.h>

// ---------------------------------------------------------------------------
// CDNA5 (gfx1250) implementation of EstNUNet.
// Pointwise/up/down convs -> bf16 WMMA GEMM (v_wmma_f32_16x16x32_bf16, f32 acc)
// Depthwise 7x7, 3x3 convs, curvature -> direct fp32 kernels.
// ---------------------------------------------------------------------------

typedef __attribute__((ext_vector_type(16))) __bf16 v16bf;
typedef __attribute__((ext_vector_type(8)))  float  v8f;

__device__ __forceinline__ unsigned short f2bf(float f) {
  unsigned int u = __float_as_uint(f);
  u += 0x7fffu + ((u >> 16) & 1u);   // round-to-nearest-even to bf16
  return (unsigned short)(u >> 16);
}

union Frag { v16bf v; uint4 q[2]; };

// ---------------------------------------------------------------------------
// WMMA GEMM: C[M,N] = A[M,K] * B[K,N]  (+ residual) (optional ReLU)
// A fp32 (row-major lda, or transposed: element (m,k) at k*lda + m)
// B fp32 row-major ldb == N. f32->bf16 conversion on the LDS-staging path.
// Block = 128 threads = 4 waves in a 2x2 grid; block tile 64x64;
// each wave computes a 32x32 tile = 4 accumulators, 4 WMMAs per K-step.
// LDS: A as [m][k], B transposed as [n][k] so every 16x32 fragment is two
// contiguous 16-byte runs -> 2x ds_load_b128 per fragment.
// Requires M,N multiples of 64 and K multiple of 32 (true for all calls).
// ---------------------------------------------------------------------------
__global__ __launch_bounds__(128)
void gemm_wmma(const float* __restrict__ A, const float* __restrict__ B,
               float* __restrict__ C, const float* __restrict__ R,
               int M, int N, int K, int lda, int transA, int relu)
{
  __shared__ unsigned short sA[64][40];  // [m][k] bf16 bits (pitch 40 -> 16B-aligned rows)
  __shared__ unsigned short sB[64][40];  // [n][k] bf16 bits (transposed)

  const int bm   = blockIdx.y * 64;
  const int bn   = blockIdx.x * 64;
  const int tid  = threadIdx.x;
  const int lane = tid & 31;
  const int wave = tid >> 5;
  const int wy   = wave >> 1;   // wave M index (0..1)
  const int wx   = wave & 1;    // wave N index (0..1)

  const int half = lane >> 4;
  const int mr   = lane & 15;
  const int ko   = half * 8;    // k-offset of fragment elements 0..7 (ushorts)

  v8f acc00 = {}, acc01 = {}, acc10 = {}, acc11 = {};

  for (int k0 = 0; k0 < K; k0 += 32) {
    // ---- stage A tile: 64(m) x 32(k) ----
#pragma unroll
    for (int it = 0; it < 4; ++it) {
      int i = tid * 4 + it * 512;           // element index, 4 consecutive k
      int r = i >> 5, c = i & 31;
      unsigned short h0, h1, h2, h3;
      if (!transA) {
        const float4 v = *(const float4*)(A + (size_t)(bm + r) * lda + (k0 + c));
        h0 = f2bf(v.x); h1 = f2bf(v.y); h2 = f2bf(v.z); h3 = f2bf(v.w);
      } else {
        h0 = f2bf(A[(size_t)(k0 + c + 0) * lda + (bm + r)]);
        h1 = f2bf(A[(size_t)(k0 + c + 1) * lda + (bm + r)]);
        h2 = f2bf(A[(size_t)(k0 + c + 2) * lda + (bm + r)]);
        h3 = f2bf(A[(size_t)(k0 + c + 3) * lda + (bm + r)]);
      }
      uint2 p;
      p.x = (unsigned int)h0 | ((unsigned int)h1 << 16);
      p.y = (unsigned int)h2 | ((unsigned int)h3 << 16);
      *(uint2*)&sA[r][c] = p;               // c multiple of 4 -> 8B aligned
    }
    // ---- stage B tile: 32(k) x 64(n), stored transposed [n][k] ----
#pragma unroll
    for (int it = 0; it < 4; ++it) {
      int i = tid * 4 + it * 512;           // 4 consecutive n, same k
      int r = i >> 6, c = i & 63;
      const float4 v = *(const float4*)(B + (size_t)(k0 + r) * N + (bn + c));
      sB[c + 0][r] = f2bf(v.x);
      sB[c + 1][r] = f2bf(v.y);
      sB[c + 2][r] = f2bf(v.z);
      sB[c + 3][r] = f2bf(v.w);
    }
    __syncthreads();

    // ---- fragments: two 16B runs each (elements 0..7 at k=ko, 8..15 at k=16+ko)
    Frag a0, a1, b0, b1;
    const unsigned short* pa0 = &sA[wy * 32 + mr][0];
    const unsigned short* pa1 = &sA[wy * 32 + 16 + mr][0];
    const unsigned short* pb0 = &sB[wx * 32 + mr][0];
    const unsigned short* pb1 = &sB[wx * 32 + 16 + mr][0];
    a0.q[0] = *(const uint4*)(pa0 + ko);
    a0.q[1] = *(const uint4*)(pa0 + 16 + ko);
    a1.q[0] = *(const uint4*)(pa1 + ko);
    a1.q[1] = *(const uint4*)(pa1 + 16 + ko);
    b0.q[0] = *(const uint4*)(pb0 + ko);
    b0.q[1] = *(const uint4*)(pb0 + 16 + ko);
    b1.q[0] = *(const uint4*)(pb1 + ko);
    b1.q[1] = *(const uint4*)(pb1 + 16 + ko);

    acc00 = __builtin_amdgcn_wmma_f32_16x16x32_bf16(false, a0.v, false, b0.v,
                                                    (short)0, acc00, false, false);
    acc01 = __builtin_amdgcn_wmma_f32_16x16x32_bf16(false, a0.v, false, b1.v,
                                                    (short)0, acc01, false, false);
    acc10 = __builtin_amdgcn_wmma_f32_16x16x32_bf16(false, a1.v, false, b0.v,
                                                    (short)0, acc10, false, false);
    acc11 = __builtin_amdgcn_wmma_f32_16x16x32_bf16(false, a1.v, false, b1.v,
                                                    (short)0, acc11, false, false);
    __syncthreads();
  }

  // ---- epilogue: C fragment rows = half*8 + r, col = lane&15 ----
  const int nn0 = bn + wx * 32 + mr;
  const int nn1 = nn0 + 16;
  const int mb0 = bm + wy * 32 + half * 8;
  const int mb1 = mb0 + 16;
#pragma unroll
  for (int r = 0; r < 8; ++r) {
    size_t i00 = (size_t)(mb0 + r) * N + nn0;
    size_t i01 = (size_t)(mb0 + r) * N + nn1;
    size_t i10 = (size_t)(mb1 + r) * N + nn0;
    size_t i11 = (size_t)(mb1 + r) * N + nn1;
    float v00 = acc00[r], v01 = acc01[r], v10 = acc10[r], v11 = acc11[r];
    if (R) { v00 += R[i00]; v01 += R[i01]; v10 += R[i10]; v11 += R[i11]; }
    if (relu) {
      v00 = fmaxf(v00, 0.f); v01 = fmaxf(v01, 0.f);
      v10 = fmaxf(v10, 0.f); v11 = fmaxf(v11, 0.f);
    }
    C[i00] = v00; C[i01] = v01; C[i10] = v10; C[i11] = v11;
  }
}

// ---------------------------------------------------------------------------
// Direct 3x3 conv, pad 1 (head: 1->64, tail: 64->2). blockIdx.z = batch.
// ---------------------------------------------------------------------------
__global__ void conv3x3_k(const float* __restrict__ x, const float* __restrict__ w,
                          float* __restrict__ y, int Cin, int Cout, int H, int W)
{
  int HW  = H * W;
  int idx = blockIdx.x * blockDim.x + threadIdx.x;
  if (idx >= Cout * HW) return;
  int n  = blockIdx.z;
  int co = idx / HW;
  int p  = idx % HW;
  int h  = p / W, wx = p % W;
  const float* xn = x + (size_t)n * Cin * HW;
  float acc = 0.f;
  for (int ci = 0; ci < Cin; ++ci) {
    const float* xc = xn + (size_t)ci * HW;
    const float* wc = w + ((size_t)co * Cin + ci) * 9;
    for (int ky = 0; ky < 3; ++ky) {
      int yy = h + ky - 1;
      if (yy < 0 || yy >= H) continue;
      for (int kx = 0; kx < 3; ++kx) {
        int xx = wx + kx - 1;
        if (xx < 0 || xx >= W) continue;
        acc += xc[yy * W + xx] * wc[ky * 3 + kx];
      }
    }
  }
  y[(size_t)n * Cout * HW + idx] = acc;
}

// ---------------------------------------------------------------------------
// Depthwise 7x7, pad 3. blockIdx.z = batch.
// ---------------------------------------------------------------------------
__global__ void dwconv7_k(const float* __restrict__ x, const float* __restrict__ w,
                          float* __restrict__ y, int C, int H, int W)
{
  int HW  = H * W;
  int idx = blockIdx.x * blockDim.x + threadIdx.x;
  if (idx >= C * HW) return;
  int n = blockIdx.z;
  int c = idx / HW;
  int p = idx % HW;
  int h = p / W, wx = p % W;
  const float* xc = x + ((size_t)n * C + c) * HW;
  const float* wc = w + (size_t)c * 49;
  float acc = 0.f;
  for (int ky = 0; ky < 7; ++ky) {
    int yy = h + ky - 3;
    if (yy < 0 || yy >= H) continue;
    for (int kx = 0; kx < 7; ++kx) {
      int xx = wx + kx - 3;
      if (xx < 0 || xx >= W) continue;
      acc += xc[yy * W + xx] * wc[ky * 7 + kx];
    }
  }
  y[((size_t)n * C + c) * HW + p] = acc;
}

// ---------------------------------------------------------------------------
// space-to-depth for stride-2 2x2 downsample conv (per image).
// Bmat[ci*4 + a*2 + b][oh*Wo+ow] = x[ci][2oh+a][2ow+b]
// ---------------------------------------------------------------------------
__global__ void s2d_k(const float* __restrict__ x, float* __restrict__ bm,
                      int Ci, int H, int W)
{
  int Ho = H >> 1, Wo = W >> 1;
  int N2 = Ho * Wo;
  int idx = blockIdx.x * blockDim.x + threadIdx.x;
  if (idx >= Ci * 4 * N2) return;
  int k = idx / N2, col = idx % N2;
  int ci = k >> 2, a = (k >> 1) & 1, b = k & 1;
  int oh = col / Wo, ow = col % Wo;
  bm[idx] = x[((size_t)ci * H + (2 * oh + a)) * W + (2 * ow + b)];
}

// ---------------------------------------------------------------------------
// pixel-shuffle + skip add for 2x2 transposed-conv upsample (per image).
// ---------------------------------------------------------------------------
__global__ void shuffle_add_k(const float* __restrict__ U, float* __restrict__ out,
                              int Co, int Hl, int Wl)
{
  int Ho = Hl * 2, Wo = Wl * 2;
  int idx = blockIdx.x * blockDim.x + threadIdx.x;
  if (idx >= Co * Ho * Wo) return;
  int c = idx / (Ho * Wo);
  int p = idx % (Ho * Wo);
  int oh = p / Wo, ow = p % Wo;
  int m = c * 4 + (oh & 1) * 2 + (ow & 1);
  out[idx] += U[(size_t)m * (Hl * Wl) + (oh >> 1) * Wl + (ow >> 1)];
}

// ---------------------------------------------------------------------------
// Post-processing
// ---------------------------------------------------------------------------
__global__ void mean_k(const float* __restrict__ tail, float* __restrict__ mean, int HW)
{
  __shared__ float red[256];
  int n = blockIdx.x;
  const float* p = tail + (size_t)n * 2 * HW;   // channel 0
  float s = 0.f;
  for (int i = threadIdx.x; i < HW; i += 256) s += p[i];
  red[threadIdx.x] = s;
  __syncthreads();
  for (int off = 128; off > 0; off >>= 1) {
    if (threadIdx.x < off) red[threadIdx.x] += red[threadIdx.x + off];
    __syncthreads();
  }
  if (threadIdx.x == 0) mean[n] = red[0] / (float)HW;
}

__global__ void assemble_k(const float* __restrict__ tail, const float* __restrict__ mean,
                           const float* __restrict__ x0, float* __restrict__ out,
                           float* __restrict__ uE, int HW, int S)
{
  int i = blockIdx.x * blockDim.x + threadIdx.x;
  if (i >= S) return;
  int n = i / HW, j = i % HW;
  out[i] = mean[n];                               // img_NL (broadcast mean)
  float nd = tail[((size_t)n * 2 + 1) * HW + j];  // channel 1
  out[S + i] = nd;                                // img_ND
  uE[i] = nd + x0[i];                             // img_E
}

__device__ __forceinline__ float gradx(const float* u, int h, int w, int W) {
  const float* r = u + (size_t)h * W;
  if (w == 0)     return r[1] - r[0];
  if (w == W - 1) return r[W - 1] - r[W - 2];
  return 0.5f * (r[w + 1] - r[w - 1]);
}
__device__ __forceinline__ float grady(const float* u, int h, int w, int H, int W) {
  if (h == 0)     return u[(size_t)W + w] - u[w];
  if (h == H - 1) return u[(size_t)(H - 1) * W + w] - u[(size_t)(H - 2) * W + w];
  return 0.5f * (u[(size_t)(h + 1) * W + w] - u[(size_t)(h - 1) * W + w]);
}

__global__ void grad_k(const float* __restrict__ uE, float* __restrict__ gx,
                       float* __restrict__ gy, int H, int W)
{
  int HW = H * W;
  int i = blockIdx.x * blockDim.x + threadIdx.x;
  if (i >= HW) return;
  int n = blockIdx.z;
  const float* u = uE + (size_t)n * HW;
  int h = i / W, w = i % W;
  gx[(size_t)n * HW + i] = gradx(u, h, w, W);
  gy[(size_t)n * HW + i] = grady(u, h, w, H, W);
}

__global__ void curv_k(const float* __restrict__ gx, const float* __restrict__ gy,
                       float* __restrict__ out, int H, int W, int S)
{
  int HW = H * W;
  int i = blockIdx.x * blockDim.x + threadIdx.x;
  if (i >= HW) return;
  int n = blockIdx.z;
  const float* uxn = gx + (size_t)n * HW;
  const float* uyn = gy + (size_t)n * HW;
  int h = i / W, w = i % W;
  float uxx = gradx(uxn, h, w, W);
  float uyy = grady(uyn, h, w, H, W);
  float uxy = grady(uxn, h, w, H, W);
  float ux = uxn[i], uy = uyn[i];
  float den = 1.f + ux * ux + uy * uy;
  out[(size_t)2 * S + (size_t)n * HW + i] = (uxx * uyy - uxy * uxy) / (den * den);
}

// ---------------------------------------------------------------------------
// Host-side orchestration helpers
// ---------------------------------------------------------------------------
static void run_stage(const float* dw, const float* w1, const float* w2,
                      float* x, float* hb, float* tb,
                      int d, int H, int W, hipStream_t s)
{
  int HW = H * W;
  for (int blk = 0; blk < 2; ++blk) {
    const float* dwi = dw + (size_t)blk * d * 49;
    const float* w1i = w1 + (size_t)blk * 4 * d * d;
    const float* w2i = w2 + (size_t)blk * 4 * d * d;
    int total = d * HW;
    dwconv7_k<<<dim3((total + 255) / 256, 1, 8), 256, 0, s>>>(x, dwi, hb, d, H, W);
    for (int n = 0; n < 8; ++n) {
      const float* Bn = hb + (size_t)n * d * HW;
      float*       xn = x  + (size_t)n * d * HW;
      // t = relu(W1 @ h)
      gemm_wmma<<<dim3(HW / 64, (4 * d) / 64), 128, 0, s>>>(
          w1i, Bn, tb, nullptr, 4 * d, HW, d, d, 0, 1);
      // x = x + W2 @ t   (residual fused, in-place)
      gemm_wmma<<<dim3(HW / 64, d / 64), 128, 0, s>>>(
          w2i, tb, xn, xn, d, HW, 4 * d, 4 * d, 0, 0);
    }
  }
}

static void run_down(const float* w, const float* xin, float* xout, float* pb,
                     int ci, int co, int H, int W, hipStream_t s)
{
  int Ho = H / 2, Wo = W / 2, N2 = Ho * Wo, K2 = ci * 4;
  for (int n = 0; n < 8; ++n) {
    int total = K2 * N2;
    s2d_k<<<dim3((total + 255) / 256), 256, 0, s>>>(
        xin + (size_t)n * ci * H * W, pb, ci, H, W);
    gemm_wmma<<<dim3(N2 / 64, co / 64), 128, 0, s>>>(
        w, pb, xout + (size_t)n * co * N2, nullptr, co, N2, K2, K2, 0, 0);
  }
}

static void run_up(const float* w, const float* xlow, float* skipx, float* pb,
                   int ci, int co, int Hl, int Wl, hipStream_t s)
{
  int Nl = Hl * Wl;
  for (int n = 0; n < 8; ++n) {
    // U[4co x Nl] = A (transposed in memory, lda = 4*co) times x_low
    gemm_wmma<<<dim3(Nl / 64, (4 * co) / 64), 128, 0, s>>>(
        w, xlow + (size_t)n * ci * Nl, pb, nullptr, 4 * co, Nl, ci, 4 * co, 1, 0);
    int total = co * 4 * Nl;
    shuffle_add_k<<<dim3((total + 255) / 256), 256, 0, s>>>(
        pb, skipx + (size_t)n * co * 4 * Nl, co, Hl, Wl);
  }
}

// ---------------------------------------------------------------------------
extern "C" void kernel_launch(void* const* d_in, const int* in_sizes, int n_in,
                              void* d_out, int out_size, void* d_ws, size_t ws_size,
                              hipStream_t stream)
{
  (void)in_sizes; (void)n_in; (void)out_size; (void)ws_size;

  const float* x0      = (const float*)d_in[0];
  const float* w_head  = (const float*)d_in[1];
  const float* en1_dw  = (const float*)d_in[2];
  const float* en1_w1  = (const float*)d_in[3];
  const float* en1_w2  = (const float*)d_in[4];
  const float* w_down1 = (const float*)d_in[5];
  const float* en2_dw  = (const float*)d_in[6];
  const float* en2_w1  = (const float*)d_in[7];
  const float* en2_w2  = (const float*)d_in[8];
  const float* w_down2 = (const float*)d_in[9];
  const float* en3_dw  = (const float*)d_in[10];
  const float* en3_w1  = (const float*)d_in[11];
  const float* en3_w2  = (const float*)d_in[12];
  const float* w_down3 = (const float*)d_in[13];
  const float* body_dw = (const float*)d_in[14];
  const float* body_w1 = (const float*)d_in[15];
  const float* body_w2 = (const float*)d_in[16];
  const float* w_up3   = (const float*)d_in[17];
  const float* de3_dw  = (const float*)d_in[18];
  const float* de3_w1  = (const float*)d_in[19];
  const float* de3_w2  = (const float*)d_in[20];
  const float* w_up2   = (const float*)d_in[21];
  const float* de2_dw  = (const float*)d_in[22];
  const float* de2_w1  = (const float*)d_in[23];
  const float* de2_w2  = (const float*)d_in[24];
  const float* w_up1   = (const float*)d_in[25];
  const float* de1_dw  = (const float*)d_in[26];
  const float* de1_w1  = (const float*)d_in[27];
  const float* de1_w2  = (const float*)d_in[28];
  const float* w_tail  = (const float*)d_in[29];

  float* out = (float*)d_out;

  // bump allocator over workspace (floats), 256B-aligned chunks
  float* ws = (float*)d_ws;
  size_t off = 0;
  auto balloc = [&](size_t nf) {
    float* p = ws + off;
    off += (nf + 63) & ~(size_t)63;
    return p;
  };
  const size_t S1 = (size_t)8 * 64 * 256 * 256;   // 33,554,432
  const size_t S2 = (size_t)8 * 128 * 128 * 128;  // 16,777,216
  const size_t S3 = (size_t)8 * 256 * 64 * 64;    //  8,388,608
  const size_t S4 = (size_t)8 * 512 * 32 * 32;    //  4,194,304

  float* a1 = balloc(S1);            // 64ch @256^2 (also en1 skip)
  float* a2 = balloc(S2);            // 128ch @128^2 (also en2 skip)
  float* a3 = balloc(S3);            // 256ch @64^2 (also en3 skip)
  float* a4 = balloc(S4);            // 512ch @32^2
  float* hb = balloc(S1);            // depthwise output (max size)
  float* tb = balloc((size_t)4 * 64 * 65536);  // per-image expand buffer (max 4d*HW)
  float* pb = balloc((size_t)256 * 16384);     // s2d / upsample U buffer
  float* tl = balloc((size_t)8 * 2 * 65536);   // tail conv output
  float* uE = balloc((size_t)8 * 65536);
  float* gx = balloc((size_t)8 * 65536);
  float* gy = balloc((size_t)8 * 65536);
  float* mn = balloc(64);

  // ---- encoder ----
  {
    int total = 64 * 256 * 256;
    conv3x3_k<<<dim3((total + 255) / 256, 1, 8), 256, 0, stream>>>(
        x0, w_head, a1, 1, 64, 256, 256);
  }
  run_stage(en1_dw, en1_w1, en1_w2, a1, hb, tb, 64, 256, 256, stream);
  run_down(w_down1, a1, a2, pb, 64, 128, 256, 256, stream);
  run_stage(en2_dw, en2_w1, en2_w2, a2, hb, tb, 128, 128, 128, stream);
  run_down(w_down2, a2, a3, pb, 128, 256, 128, 128, stream);
  run_stage(en3_dw, en3_w1, en3_w2, a3, hb, tb, 256, 64, 64, stream);
  run_down(w_down3, a3, a4, pb, 256, 512, 64, 64, stream);

  // ---- body ----
  run_stage(body_dw, body_w1, body_w2, a4, hb, tb, 512, 32, 32, stream);

  // ---- decoder ----
  run_up(w_up3, a4, a3, pb, 512, 256, 32, 32, stream);    // a3 = up(a4) + en3 skip
  run_stage(de3_dw, de3_w1, de3_w2, a3, hb, tb, 256, 64, 64, stream);
  run_up(w_up2, a3, a2, pb, 256, 128, 64, 64, stream);
  run_stage(de2_dw, de2_w1, de2_w2, a2, hb, tb, 128, 128, 128, stream);
  run_up(w_up1, a2, a1, pb, 128, 64, 128, 128, stream);
  run_stage(de1_dw, de1_w1, de1_w2, a1, hb, tb, 64, 256, 256, stream);

  // ---- tail ----
  {
    int total = 2 * 256 * 256;
    conv3x3_k<<<dim3((total + 255) / 256, 1, 8), 256, 0, stream>>>(
        a1, w_tail, tl, 64, 2, 256, 256);
  }

  // ---- post-process ----
  const int HW = 256 * 256;
  const int S  = 8 * HW;
  mean_k<<<dim3(8), 256, 0, stream>>>(tl, mn, HW);
  assemble_k<<<dim3((S + 255) / 256), 256, 0, stream>>>(tl, mn, x0, out, uE, HW, S);
  grad_k<<<dim3((HW + 255) / 256, 1, 8), 256, 0, stream>>>(uE, gx, gy, 256, 256);
  curv_k<<<dim3((HW + 255) / 256, 1, 8), 256, 0, stream>>>(gx, gy, out, 256, 256, S);
}
